// CausalSelfAttention_48249662603766
// MI455X (gfx1250) — matrix operand order repaired and checked by reading, not compiled, not run on hardware
//
#include <hip/hip_runtime.h>

// CausalSelfAttention for MI455X (gfx1250, wave32, WMMA f16->f32).
// M=8, B=4, S=512, D=512, H=8, HD=64.
// Compute-bound (~69 GFLOP vs ~100MB traffic) -> all GEMMs via
// v_wmma_f32_16x16x32_f16. Working set fits 192MB L2, so fragments are
// loaded directly from global memory as 16B-contiguous b128 chunks.

static constexpr int MD = 8;     // models
static constexpr int BD = 4;     // batch
static constexpr int SD = 512;   // sequence
static constexpr int DD = 512;   // model dim
static constexpr int HD_ = 8;    // heads
static constexpr int HDD = 64;   // head dim

typedef __attribute__((ext_vector_type(16))) _Float16 v16h;
typedef __attribute__((ext_vector_type(8)))  _Float16 v8h;
typedef __attribute__((ext_vector_type(8)))  float    v8f;

// ---- WMMA fragment helpers (layouts per CDNA5 ISA §7.12.2) ----

// A-matrix 16x32 f16 from row-major [row][k] data, 16B-aligned runs.
// lane l: row = l&15, runs of 8 halves at k = (l>>4)*8 and 16+(l>>4)*8.
static __device__ __forceinline__ v16h load_a_frag(const _Float16* base, int lda) {
  const int lane = threadIdx.x & 31;
  const _Float16* p = base + (lane & 15) * lda + ((lane >> 4) << 3);
  v8h lo = *reinterpret_cast<const v8h*>(p);
  v8h hi = *reinterpret_cast<const v8h*>(p + 16);
  v16h r;
#pragma unroll
  for (int i = 0; i < 8; ++i) { r[i] = lo[i]; r[i + 8] = hi[i]; }
  return r;
}

// B-matrix 32x16 f16 from [n][k]-major data (k contiguous).
// lane l: col = l&15, runs of 8 halves at k = (l>>4)*8 and 16+(l>>4)*8.
static __device__ __forceinline__ v16h load_b_frag(const _Float16* base, int ldk) {
  const int lane = threadIdx.x & 31;
  const _Float16* p = base + (lane & 15) * ldk + ((lane >> 4) << 3);
  v8h lo = *reinterpret_cast<const v8h*>(p);
  v8h hi = *reinterpret_cast<const v8h*>(p + 16);
  v16h r;
#pragma unroll
  for (int i = 0; i < 8; ++i) { r[i] = lo[i]; r[i + 8] = hi[i]; }
  return r;
}

static __device__ __forceinline__ v8f wmma16(v16h a, v16h b, v8f c) {
  return __builtin_amdgcn_wmma_f32_16x16x32_f16(false, a, false, b, (short)0, c,
                                                false, false);
}

// ---- one-time precision/layout conversion kernels ----

__global__ void k_cvt_f16(const float* __restrict__ in, _Float16* __restrict__ out,
                          long n) {
  long i = (long)blockIdx.x * blockDim.x + threadIdx.x;
  if (i < n) out[i] = (_Float16)in[i];
}

// in[m][k][n] f32  ->  out[m][n][k] f16   (batch m = blockIdx.y)
__global__ void k_transpose_cvt(const float* __restrict__ in,
                                _Float16* __restrict__ out, int K, int N) {
  long idx = (long)blockIdx.x * blockDim.x + threadIdx.x;
  if (idx >= (long)K * N) return;
  int k = (int)(idx / N);
  int n = (int)(idx % N);
  long mb = (long)blockIdx.y * K * N;
  out[mb + (long)n * K + k] = (_Float16)in[mb + idx];
}

// ---- QKV projection GEMM: [2048 x 512] @ [512 x 1536] per model ----
// Writes Q [m][b][h][s][hd], K [m][b][h][s][hd], V^T [m][b][h][hd][s] in f16.
__global__ __launch_bounds__(256) void k_qkv_gemm(
    const _Float16* __restrict__ x16, const _Float16* __restrict__ wT,
    _Float16* __restrict__ qws, _Float16* __restrict__ kws,
    _Float16* __restrict__ vT) {
  const int m  = blockIdx.z;
  const int r0 = blockIdx.y * 128 + ((threadIdx.x >> 5) >> 2) * 64;
  const int c0 = blockIdx.x * 128 + ((threadIdx.x >> 5) & 3) * 32;
  const int lane = threadIdx.x & 31;

  const _Float16* A  = x16 + ((long)m * BD * SD + r0) * DD;
  const _Float16* Bm = wT + ((long)m * 3 * DD + c0) * DD;

  v8f acc[4][2];
#pragma unroll
  for (int i = 0; i < 4; ++i)
#pragma unroll
    for (int j = 0; j < 2; ++j) acc[i][j] = {};

  for (int k0 = 0; k0 < DD; k0 += 32) {
    v16h a[4], b[2];
#pragma unroll
    for (int i = 0; i < 4; ++i) a[i] = load_a_frag(A + (long)i * 16 * DD + k0, DD);
#pragma unroll
    for (int j = 0; j < 2; ++j) b[j] = load_b_frag(Bm + (long)j * 16 * DD + k0, DD);
#pragma unroll
    for (int i = 0; i < 4; ++i)
#pragma unroll
      for (int j = 0; j < 2; ++j) acc[i][j] = wmma16(a[i], b[j], acc[i][j]);
  }

  const int colL = lane & 15;
  const int rowH = (lane >> 4) * 8;
#pragma unroll
  for (int i = 0; i < 4; ++i) {
#pragma unroll
    for (int j = 0; j < 2; ++j) {
      const int n = c0 + j * 16 + colL;
      const int chunk = n >> 9;          // 0=q 1=k 2=v
      const int d = n & 511;
      const int h = d >> 6, hd = d & 63;
#pragma unroll
      for (int ii = 0; ii < 8; ++ii) {
        const int r = r0 + i * 16 + rowH + ii;
        const int bi = r >> 9, s = r & 511;
        const long head = ((long)m * BD + bi) * HD_ + h;
        const _Float16 v = (_Float16)acc[i][j][ii];
        if (chunk == 0)      qws[head * SD * HDD + (long)s * HDD + hd] = v;
        else if (chunk == 1) kws[head * SD * HDD + (long)s * HDD + hd] = v;
        else                 vT[head * HDD * SD + (long)hd * SD + s] = v;
      }
    }
  }
}

// ---- Flash attention: one (head, 128-q-row block) per workgroup ----
__global__ __launch_bounds__(256) void k_attn(
    const _Float16* __restrict__ qws, const _Float16* __restrict__ kws,
    const _Float16* __restrict__ vT, _Float16* __restrict__ yws) {
  __shared__ _Float16 lds[8 * 16 * 40];   // per-wave 16x32 P tile, pad->40
  const int head = blockIdx.x;            // (m*B + b)*H + h
  const int wv   = threadIdx.x >> 5;
  const int lane = threadIdx.x & 31;
  const int q0   = blockIdx.y * 128 + wv * 16;
  const int colL = lane & 15;
  const int rowH = (lane >> 4) * 8;

  const _Float16* Q = qws + (long)head * SD * HDD;
  const _Float16* K = kws + (long)head * SD * HDD;
  const _Float16* V = vT  + (long)head * HDD * SD;
  _Float16* ldsw = lds + wv * 16 * 40;

  v16h qf[2];
  qf[0] = load_a_frag(Q + (long)q0 * HDD + 0,  HDD);
  qf[1] = load_a_frag(Q + (long)q0 * HDD + 32, HDD);

  float mrow[8], lrow[8];
  v8f o[4];
#pragma unroll
  for (int i = 0; i < 8; ++i) { mrow[i] = -__builtin_inff(); lrow[i] = 0.f; }
#pragma unroll
  for (int nt = 0; nt < 4; ++nt) o[nt] = {};

  const float scale = 0.125f;  // 1/sqrt(64)

  for (int j0 = 0; j0 <= q0 + 15; j0 += 32) {
    v8f s0 = {}, s1 = {};
#pragma unroll
    for (int kk = 0; kk < 2; ++kk) {
      v16h b0 = load_b_frag(K + (long)j0 * HDD + kk * 32, HDD);
      v16h b1 = load_b_frag(K + (long)(j0 + 16) * HDD + kk * 32, HDD);
      s0 = wmma16(qf[kk], b0, s0);
      s1 = wmma16(qf[kk], b1, s1);
    }
    // online softmax over this 16x32 score slab (rows live per lane-half)
#pragma unroll
    for (int i = 0; i < 8; ++i) {
      const int qrow = q0 + rowH + i;
      const int key0 = j0 + colL;
      float v0 = (key0 <= qrow)      ? s0[i] * scale : -__builtin_inff();
      float v1 = (key0 + 16 <= qrow) ? s1[i] * scale : -__builtin_inff();
      float mx = fmaxf(v0, v1);
#pragma unroll
      for (int dd = 1; dd < 16; dd <<= 1) mx = fmaxf(mx, __shfl_xor(mx, dd, 16));
      const float mnew  = fmaxf(mrow[i], mx);
      const float alpha = __expf(mrow[i] - mnew);
      const float p0 = __expf(v0 - mnew);
      const float p1 = __expf(v1 - mnew);
      float sum = p0 + p1;
#pragma unroll
      for (int dd = 1; dd < 16; dd <<= 1) sum += __shfl_xor(sum, dd, 16);
      lrow[i] = lrow[i] * alpha + sum;
      mrow[i] = mnew;
#pragma unroll
      for (int nt = 0; nt < 4; ++nt) o[nt][i] *= alpha;
      ldsw[(rowH + i) * 40 + colL]      = (_Float16)p0;
      ldsw[(rowH + i) * 40 + 16 + colL] = (_Float16)p1;
    }
    asm volatile("s_wait_dscnt 0" ::: "memory");  // wave-private LDS RAW
    v16h pa = load_a_frag(ldsw, 40);              // P as 16x32 A-fragment
#pragma unroll
    for (int nt = 0; nt < 4; ++nt) {
      v16h bv = load_b_frag(V + (long)(nt * 16) * SD + j0, SD);
      o[nt] = wmma16(pa, bv, o[nt]);
    }
  }

  const long mb = head / HD_;
  const int  h  = head % HD_;
#pragma unroll
  for (int i = 0; i < 8; ++i) {
    const float inv = 1.0f / lrow[i];
    const int s = q0 + rowH + i;
#pragma unroll
    for (int nt = 0; nt < 4; ++nt)
      yws[(mb * SD + s) * (long)DD + h * HDD + nt * 16 + colL] =
          (_Float16)(o[nt][i] * inv);
  }
}

// ---- Output projection GEMM: [2048 x 512] @ [512 x 512] per model ----
__global__ __launch_bounds__(256) void k_proj(
    const _Float16* __restrict__ yws, const _Float16* __restrict__ pT,
    float* __restrict__ out) {
  const int m  = blockIdx.z;
  const int r0 = blockIdx.y * 128 + ((threadIdx.x >> 5) >> 2) * 64;
  const int c0 = blockIdx.x * 128 + ((threadIdx.x >> 5) & 3) * 32;
  const int lane = threadIdx.x & 31;

  const _Float16* A  = yws + ((long)m * BD * SD + r0) * DD;
  const _Float16* Bm = pT + ((long)m * DD + c0) * DD;

  v8f acc[4][2];
#pragma unroll
  for (int i = 0; i < 4; ++i)
#pragma unroll
    for (int j = 0; j < 2; ++j) acc[i][j] = {};

  for (int k0 = 0; k0 < DD; k0 += 32) {
    v16h a[4], b[2];
#pragma unroll
    for (int i = 0; i < 4; ++i) a[i] = load_a_frag(A + (long)i * 16 * DD + k0, DD);
#pragma unroll
    for (int j = 0; j < 2; ++j) b[j] = load_b_frag(Bm + (long)j * 16 * DD + k0, DD);
#pragma unroll
    for (int i = 0; i < 4; ++i)
#pragma unroll
      for (int j = 0; j < 2; ++j) acc[i][j] = wmma16(a[i], b[j], acc[i][j]);
  }

  const int colL = lane & 15;
  const int rowH = (lane >> 4) * 8;
#pragma unroll
  for (int i = 0; i < 4; ++i)
#pragma unroll
    for (int j = 0; j < 2; ++j) {
      const int n = c0 + j * 16 + colL;
#pragma unroll
      for (int ii = 0; ii < 8; ++ii) {
        const int r = r0 + i * 16 + rowH + ii;
        out[((long)m * BD * SD + r) * DD + n] = acc[i][j][ii];
      }
    }
}

extern "C" void kernel_launch(void* const* d_in, const int* in_sizes, int n_in,
                              void* d_out, int out_size, void* d_ws, size_t ws_size,
                              hipStream_t stream) {
  const float* x     = (const float*)d_in[0];
  const float* qkvw  = (const float*)d_in[1];
  const float* projw = (const float*)d_in[2];
  float* out = (float*)d_out;

  char* ws = (char*)d_ws;
  size_t off = 0;
  const size_t szX   = (size_t)MD * BD * SD * DD * sizeof(_Float16);   // 16.8 MB
  const size_t szQkv = (size_t)MD * 3 * DD * DD * sizeof(_Float16);    // 12.6 MB
  const size_t szPrj = (size_t)MD * DD * DD * sizeof(_Float16);        //  4.2 MB
  _Float16* x16   = (_Float16*)(ws + off); off += szX;
  _Float16* qkvT  = (_Float16*)(ws + off); off += szQkv;
  _Float16* projT = (_Float16*)(ws + off); off += szPrj;
  _Float16* qws   = (_Float16*)(ws + off); off += szX;
  _Float16* kws   = (_Float16*)(ws + off); off += szX;
  _Float16* vT    = (_Float16*)(ws + off); off += szX;
  _Float16* yws   = x16;  // live ranges don't overlap: reuse x16 buffer

  const long nx = (long)MD * BD * SD * DD;
  k_cvt_f16<<<(unsigned)((nx + 255) / 256), 256, 0, stream>>>(x, x16, nx);

  dim3 gtq((unsigned)(((long)DD * 3 * DD + 255) / 256), MD);
  k_transpose_cvt<<<gtq, 256, 0, stream>>>(qkvw, qkvT, DD, 3 * DD);
  dim3 gtp((unsigned)(((long)DD * DD + 255) / 256), MD);
  k_transpose_cvt<<<gtp, 256, 0, stream>>>(projw, projT, DD, DD);

  dim3 gq(12, 16, MD);               // 1536/128 cols, 2048/128 rows, m
  k_qkv_gemm<<<gq, 256, 0, stream>>>(x16, qkvT, qws, kws, vT);

  dim3 ga(MD * BD * HD_, SD / 128);  // 256 heads x 4 q-blocks
  k_attn<<<ga, 256, 0, stream>>>(qws, kws, vT, yws);

  dim3 gp(4, 16, MD);                // 512/128 cols, 2048/128 rows, m
  k_proj<<<gp, 256, 0, stream>>>(yws, projT, out);
}
